// AttnBlock3D_27977416966625
// MI455X (gfx1250) — matrix operand order repaired and checked by reading, not compiled, not run on hardware
//
#include <hip/hip_runtime.h>
#include <stdint.h>

typedef __attribute__((ext_vector_type(16))) __bf16 v16bf;
typedef __attribute__((ext_vector_type(8)))  float  v8f;

#define B_   2
#define C_   512
#define N_   4096
#define G_   32
#define CPG  16
#define QSCALE 0.04419417382415922f   /* 512^-0.5 */

union Frag { uint4 q[2]; v16bf v; };

static __device__ __forceinline__ unsigned short f2bf(float f) {
  unsigned u = __float_as_uint(f);
  u += 0x7FFFu + ((u >> 16) & 1u);       // round to nearest even
  return (unsigned short)(u >> 16);
}

static __device__ __forceinline__ v8f wmma_bf16(const Frag& a, const Frag& b, v8f c) {
  return __builtin_amdgcn_wmma_f32_16x16x32_bf16(
      /*neg_a=*/false, a.v, /*neg_b=*/false, b.v,
      /*c_mod=*/(short)0, c, /*reuse_a=*/false, /*reuse_b=*/false);
}

// ---------------------------------------------------------------- GroupNorm
__global__ __launch_bounds__(256)
void k_gn_stats(const float* __restrict__ x, float* __restrict__ stats) {
  const int bg = blockIdx.x;                       // b*32 + g
  const float* base = x + (size_t)bg * CPG * N_;   // channels are group-contiguous
  float s = 0.f, ss = 0.f;
  for (int i = threadIdx.x; i < CPG * N_; i += 256) {
    float v = base[i];
    s += v; ss += v * v;
  }
  __shared__ float rs[256], rq[256];
  rs[threadIdx.x] = s; rq[threadIdx.x] = ss;
  __syncthreads();
  for (int o = 128; o > 0; o >>= 1) {
    if (threadIdx.x < o) { rs[threadIdx.x] += rs[threadIdx.x + o];
                           rq[threadIdx.x] += rq[threadIdx.x + o]; }
    __syncthreads();
  }
  if (threadIdx.x == 0) {
    const float inv = 1.0f / (float)(CPG * N_);
    float m = rs[0] * inv;
    float var = rq[0] * inv - m * m;
    stats[bg * 2 + 0] = m;
    stats[bg * 2 + 1] = rsqrtf(var + 1e-6f);
  }
}

__global__ __launch_bounds__(256)
void k_gn_apply(const float* __restrict__ x, const float* __restrict__ stats,
                const float* __restrict__ gw, const float* __restrict__ gb,
                unsigned short* __restrict__ seq) {
  const int bc = blockIdx.x;                  // b*512 + c
  const int b = bc >> 9, c = bc & (C_ - 1);
  const int g = c / CPG;
  const float m = stats[(b * G_ + g) * 2 + 0];
  const float r = stats[(b * G_ + g) * 2 + 1];
  const float a = r * gw[c];
  const float bb = gb[c] - m * a;
  const float* xr = x + (size_t)bc * N_;
  for (int n = threadIdx.x; n < N_; n += 256)
    seq[((size_t)(b * N_ + n)) * C_ + c] = f2bf(xr[n] * a + bb);
}

// ------------------------------------------------------- weight conversion
__global__ __launch_bounds__(256)
void k_cvt_w(const float* __restrict__ qw, const float* __restrict__ pw,
             unsigned short* __restrict__ qwb, unsigned short* __restrict__ pwb) {
  const int i = blockIdx.x * 256 + threadIdx.x;
  if (i < 3 * C_ * C_) qwb[i] = f2bf(qw[i]);
  if (i < C_ * C_)     pwb[i] = f2bf(pw[i]);
}

// ------------------------------------------------------------ QKV GEMM (NT)
// seq[8192,512] x Wqkv[1536,512]^T ; split epilogue into Q (scaled), K, V^T
__global__ __launch_bounds__(256)
void k_qkv(const unsigned short* __restrict__ seq, const unsigned short* __restrict__ wb,
           const float* __restrict__ bias, unsigned short* __restrict__ qb,
           unsigned short* __restrict__ kbuf, unsigned short* __restrict__ vtb) {
  const int lane = threadIdx.x & 31, w = threadIdx.x >> 5;
  const int half = lane >> 4, l16 = lane & 15;
  const int mb = blockIdx.x * 64 + (w & 1) * 32;
  const int nb = blockIdx.y * 128 + (w >> 1) * 32;
  v8f acc[2][2];
  for (int i = 0; i < 2; i++) for (int j = 0; j < 2; j++)
    for (int e = 0; e < 8; e++) acc[i][j][e] = 0.f;
  for (int kk = 0; kk < C_; kk += 32) {
    Frag a[2], bf[2];
    for (int mi = 0; mi < 2; mi++) {
      const unsigned short* p = seq + (size_t)(mb + mi * 16 + l16) * C_ + kk + half * 8;
      a[mi].q[0] = *(const uint4*)p;
      a[mi].q[1] = *(const uint4*)(p + 16);
    }
    for (int ni = 0; ni < 2; ni++) {
      const unsigned short* p = wb + (size_t)(nb + ni * 16 + l16) * C_ + kk + half * 8;
      bf[ni].q[0] = *(const uint4*)p;
      bf[ni].q[1] = *(const uint4*)(p + 16);
    }
    for (int mi = 0; mi < 2; mi++)
      for (int ni = 0; ni < 2; ni++)
        acc[mi][ni] = wmma_bf16(a[mi], bf[ni], acc[mi][ni]);
  }
  for (int mi = 0; mi < 2; mi++)
    for (int ni = 0; ni < 2; ni++) {
      const int col = nb + ni * 16 + l16;
      const float bv = bias[col];
      for (int e = 0; e < 8; e++) {
        const int m = mb + mi * 16 + e + half * 8;
        const float v = acc[mi][ni][e] + bv;
        if (col < C_)            qb  [(size_t)m * C_ + col]        = f2bf(v * QSCALE);
        else if (col < 2 * C_)   kbuf[(size_t)m * C_ + (col - C_)] = f2bf(v);
        else {
          const int b = m >> 12, n = m & (N_ - 1);
          vtb[((size_t)(b * C_ + (col - 2 * C_))) * N_ + n] = f2bf(v);
        }
      }
    }
}

// --------------------------------------------------------- flash attention
// one 16-row q tile per workgroup; 8 waves split C=512 into 64-wide chunks.
// Wave 0 alone reduces partial S, does the online softmax and publishes
// P (bf16) + per-row alpha; other waves overlap their V-fragment loads.
__global__ __launch_bounds__(256)
void k_attn(const unsigned short* __restrict__ qb, const unsigned short* __restrict__ kbuf,
            const unsigned short* __restrict__ vtb, unsigned short* __restrict__ ob) {
  __shared__ __align__(16) float sp[8 * 512];           // partials: [w][tile][e*32+lane]
  __shared__ __align__(16) unsigned short pl[16 * 32];  // P tile (bf16, row-major)
  __shared__ float al[16];                              // alpha per row
  __shared__ float linv[16];                            // 1/l per row
  const int lane = threadIdx.x & 31, w = threadIdx.x >> 5;
  const int half = lane >> 4, l16 = lane & 15;
  const int b = blockIdx.x >> 8, qbase = (blockIdx.x & 255) * 16;
  const int cw = w * 64;

  Frag qa[2];
  for (int cc = 0; cc < 2; cc++) {
    const unsigned short* p = qb + (size_t)(b * N_ + qbase + l16) * C_ + cw + cc * 32 + half * 8;
    qa[cc].q[0] = *(const uint4*)p;
    qa[cc].q[1] = *(const uint4*)(p + 16);
  }
  v8f o[4];
  for (int ct = 0; ct < 4; ct++) for (int e = 0; e < 8; e++) o[ct][e] = 0.f;
  float mrow[8], lrow[8];                 // live only in wave 0
  for (int e = 0; e < 8; e++) { mrow[e] = -1e30f; lrow[e] = 0.f; }

  for (int kt = 0; kt < N_; kt += 32) {
    // partial S over this wave's 64 channels, two 16-col tiles
    v8f s0, s1;
    for (int e = 0; e < 8; e++) { s0[e] = 0.f; s1[e] = 0.f; }
    for (int cc = 0; cc < 2; cc++) {
      Frag k0, k1;
      const unsigned short* p0 = kbuf + (size_t)(b * N_ + kt + l16) * C_ + cw + cc * 32 + half * 8;
      k0.q[0] = *(const uint4*)p0; k0.q[1] = *(const uint4*)(p0 + 16);
      const unsigned short* p1 = kbuf + (size_t)(b * N_ + kt + 16 + l16) * C_ + cw + cc * 32 + half * 8;
      k1.q[0] = *(const uint4*)p1; k1.q[1] = *(const uint4*)(p1 + 16);
      s0 = wmma_bf16(qa[cc], k0, s0);
      s1 = wmma_bf16(qa[cc], k1, s1);
    }
    if (kt + 32 < N_)
      __builtin_prefetch(kbuf + (size_t)(b * N_ + kt + 32 + l16) * C_ + cw, 0, 0);

    // bank-conflict-free partial exchange: one element per bank per access
    float* dst = &sp[w * 512 + lane];
    for (int e = 0; e < 8; e++) { dst[e * 32] = s0[e]; dst[256 + e * 32] = s1[e]; }
    __syncthreads();                                    // barrier 1

    // hoist V-fragment loads (independent of LDS) to overlap wave 0's softmax
    Frag vf[4];
    for (int ct = 0; ct < 4; ct++) {
      const unsigned short* p = vtb + ((size_t)(b * C_ + cw + ct * 16 + l16)) * N_ + kt + half * 8;
      vf[ct].q[0] = *(const uint4*)p; vf[ct].q[1] = *(const uint4*)(p + 16);
    }

    if (w == 0) {                                       // single-wave softmax
      float f0[8], f1[8];
      for (int e = 0; e < 8; e++) { f0[e] = 0.f; f1[e] = 0.f; }
      for (int ww = 0; ww < 8; ww++) {
        const float* s = &sp[ww * 512 + lane];
        for (int e = 0; e < 8; e++) { f0[e] += s[e * 32]; f1[e] += s[256 + e * 32]; }
      }
      float alpha[8];
      for (int e = 0; e < 8; e++) {
        float mx = fmaxf(f0[e], f1[e]);
        for (int off = 8; off > 0; off >>= 1) mx = fmaxf(mx, __shfl_xor(mx, off, 32));
        const float mnew = fmaxf(mrow[e], mx);
        alpha[e] = __expf(mrow[e] - mnew);
        mrow[e] = mnew;
        f0[e] = __expf(f0[e] - mnew);
        f1[e] = __expf(f1[e] - mnew);
        float rsum = f0[e] + f1[e];
        for (int off = 8; off > 0; off >>= 1) rsum += __shfl_xor(rsum, off, 32);
        lrow[e] = lrow[e] * alpha[e] + rsum;
      }
      for (int e = 0; e < 8; e++) {                     // publish P + alpha
        const int row = e + half * 8;
        pl[row * 32 + l16]      = f2bf(f0[e]);
        pl[row * 32 + 16 + l16] = f2bf(f1[e]);
      }
      if (l16 == 0)
        for (int e = 0; e < 8; e++) al[e + half * 8] = alpha[e];
    }
    __syncthreads();                                    // barrier 2

    float av[8];
    for (int e = 0; e < 8; e++) av[e] = al[e + half * 8];   // broadcast reads
    for (int ct = 0; ct < 4; ct++)
      for (int e = 0; e < 8; e++) o[ct][e] *= av[e];

    Frag pf;
    {
      const unsigned short* p = &pl[l16 * 32 + half * 8];
      pf.q[0] = *(const uint4*)p;
      pf.q[1] = *(const uint4*)(p + 16);
    }
    for (int ct = 0; ct < 4; ct++)
      o[ct] = wmma_bf16(pf, vf[ct], o[ct]);
    // no third barrier needed: next iteration's sp/pl writes sit behind the
    // next barrier 1 / barrier 2, after all consumers of this step are done.
  }

  if (w == 0 && l16 == 0)
    for (int e = 0; e < 8; e++) linv[e + half * 8] = 1.0f / lrow[e];
  __syncthreads();
  float li[8];
  for (int e = 0; e < 8; e++) li[e] = linv[e + half * 8];
  for (int ct = 0; ct < 4; ct++) {
    const int c = cw + ct * 16 + l16;
    for (int e = 0; e < 8; e++) {
      const int row = qbase + e + half * 8;
      ob[(size_t)(b * N_ + row) * C_ + c] = f2bf(o[ct][e] * li[e]);
    }
  }
}

// ----------------------------------------------- proj GEMM + bias + residual
__global__ __launch_bounds__(256)
void k_proj(const unsigned short* __restrict__ ob, const unsigned short* __restrict__ wb,
            const float* __restrict__ bias, const float* __restrict__ x,
            float* __restrict__ out) {
  const int lane = threadIdx.x & 31, w = threadIdx.x >> 5;
  const int half = lane >> 4, l16 = lane & 15;
  const int mb = blockIdx.x * 64 + (w & 1) * 32;
  const int nb = blockIdx.y * 128 + (w >> 1) * 32;
  v8f acc[2][2];
  for (int i = 0; i < 2; i++) for (int j = 0; j < 2; j++)
    for (int e = 0; e < 8; e++) acc[i][j][e] = 0.f;
  for (int kk = 0; kk < C_; kk += 32) {
    Frag a[2], bf[2];
    for (int mi = 0; mi < 2; mi++) {
      const unsigned short* p = ob + (size_t)(mb + mi * 16 + l16) * C_ + kk + half * 8;
      a[mi].q[0] = *(const uint4*)p;
      a[mi].q[1] = *(const uint4*)(p + 16);
    }
    for (int ni = 0; ni < 2; ni++) {
      const unsigned short* p = wb + (size_t)(nb + ni * 16 + l16) * C_ + kk + half * 8;
      bf[ni].q[0] = *(const uint4*)p;
      bf[ni].q[1] = *(const uint4*)(p + 16);
    }
    for (int mi = 0; mi < 2; mi++)
      for (int ni = 0; ni < 2; ni++)
        acc[mi][ni] = wmma_bf16(a[mi], bf[ni], acc[mi][ni]);
  }
  for (int mi = 0; mi < 2; mi++)
    for (int ni = 0; ni < 2; ni++) {
      const int col = nb + ni * 16 + l16;
      const float bv = bias[col];
      for (int e = 0; e < 8; e++) {
        const int m = mb + mi * 16 + e + half * 8;
        const int b = m >> 12, n = m & (N_ - 1);
        const size_t idx = ((size_t)(b * C_ + col)) * N_ + n;
        out[idx] = x[idx] + acc[mi][ni][e] + bv;
      }
    }
}

// -------------------------------------------------------------------- launch
extern "C" void kernel_launch(void* const* d_in, const int* in_sizes, int n_in,
                              void* d_out, int out_size, void* d_ws, size_t ws_size,
                              hipStream_t stream) {
  const float* x     = (const float*)d_in[0];
  const float* gw    = (const float*)d_in[1];
  const float* gb    = (const float*)d_in[2];
  const float* qw    = (const float*)d_in[3];
  const float* qbias = (const float*)d_in[4];
  const float* pw    = (const float*)d_in[5];
  const float* pbias = (const float*)d_in[6];
  float* out = (float*)d_out;

  char* ws = (char*)d_ws;
  float* stats          = (float*)ws;                         // 128 floats
  unsigned short* seq   = (unsigned short*)(ws + 1024);       // [8192,512] bf16
  unsigned short* qb    = seq  + (size_t)8192 * 512;
  unsigned short* kbuf  = qb   + (size_t)8192 * 512;
  unsigned short* vtb   = kbuf + (size_t)8192 * 512;          // [B,512,4096]
  unsigned short* ob    = vtb  + (size_t)8192 * 512;
  unsigned short* qwb   = ob   + (size_t)8192 * 512;          // [1536,512]
  unsigned short* pwb   = qwb  + (size_t)1536 * 512;          // [512,512]

  k_gn_stats<<<64,   256, 0, stream>>>(x, stats);
  k_gn_apply<<<1024, 256, 0, stream>>>(x, stats, gw, gb, seq);
  k_cvt_w   <<<3072, 256, 0, stream>>>(qw, pw, qwb, pwb);
  k_qkv     <<<dim3(128, 12), 256, 0, stream>>>(seq, qwb, qbias, qb, kbuf, vtb);
  k_attn    <<<512,  256, 0, stream>>>(qb, kbuf, vtb, ob);
  k_proj    <<<dim3(128, 4), 256, 0, stream>>>(ob, pwb, pbias, x, out);
}
